// LSTMModel_7885559955934
// MI455X (gfx1250) — matrix operand order repaired and checked by reading, not compiled, not run on hardware
//
#include <hip/hip_runtime.h>

typedef __attribute__((ext_vector_type(16))) _Float16 v16h;
typedef __attribute__((ext_vector_type(8)))  _Float16 v8h;
typedef __attribute__((ext_vector_type(2)))  _Float16 v2h;
typedef __attribute__((ext_vector_type(2)))  __fp16   v2fp16;
typedef __attribute__((ext_vector_type(8)))  float    v8f;

#define B_TOTAL 4096
#define T_STEPS 512
#define HID     32

// gfx1250 V_TANH_F32 (probe-confirmed: lowers to v_tanh_f32_e32).
#if __has_builtin(__builtin_amdgcn_tanhf)
__device__ __forceinline__ float fast_tanh(float x) { return __builtin_amdgcn_tanhf(x); }
#else
__device__ __forceinline__ float fast_tanh(float x) {
    float e = __builtin_amdgcn_exp2f(2.885390082f * x);
    return 1.0f - 2.0f * __builtin_amdgcn_rcpf(1.0f + e);
}
#endif

// Gate pre-activations for i,f,o arrive PRE-SCALED by 0.5 (folded into weights),
// so sigmoid(g) = 0.5*tanh(g/2)+0.5 = fma(tanh(D_scaled), 0.5, 0.5).
__device__ __forceinline__ float sigmoid_from_half(float half_x) {
    return fmaf(fast_tanh(half_x), 0.5f, 0.5f);
}

// One wave (32 lanes) owns 16 batch rows for all 512 timesteps.
// Per step: gates[16,128] = h[16,32] @ Whh^T[32,128] + (x*w_ih + b)
// computed as 8x V_WMMA_F32_16X16X32_F16 (M=16 batch, K=32 hidden, N=16 gates/tile).
//
// K-permutation trick: LDS h tile stores hidden k at element row*32 + s with
// k = perm(s) = (s>>1) + 16*(s&1), making each lane's two produced h values
// (hidden lm, lm+16) adjacent -> one packed b32 store. B fragments apply the
// same K permutation, so the WMMA result is unchanged.
__global__ __launch_bounds__(32) void lstm_wmma_kernel(
    const float* __restrict__ x,     // [B, T] (I==1)
    const float* __restrict__ Wih,   // [128, 1]
    const float* __restrict__ Whh,   // [128, 32]
    const float* __restrict__ bih,   // [128]
    const float* __restrict__ bhh,   // [128]
    const float* __restrict__ Wlin,  // [1, 32]
    const float* __restrict__ blin,  // [1]
    float* __restrict__ out)         // [B]
{
    __shared__ __align__(16) _Float16 hbuf[16 * HID];  // [batch16][K-position], f16

    const int lane = threadIdx.x;        // 0..31
    const int half = lane >> 4;          // 0 | 1
    const int lm   = lane & 15;
    const int batch_base = blockIdx.x * 16;

    // ---- loop-invariant per-lane constants -------------------------------
    // Sigmoid tiles (i,f,o = n in {0,1,2,3,6,7}) are pre-scaled by 0.5 so the
    // activation needs no input multiply; tanh tiles (g = n in {4,5}) unscaled.
    float wihb[8], biasb[8];
    v16h bfrag[8];   // lane = N col (gate); lanes<16: K-pos 0..7,16..23, lanes>=16: 8..15,24..31
#pragma unroll
    for (int n = 0; n < 8; ++n) {
        const float scale = (n == 4 || n == 5) ? 1.0f : 0.5f;  // exact (2^-1)
        const int g = 16 * n + lm;
        wihb[n]  = Wih[g] * scale;
        biasb[n] = (bih[g] + bhh[g]) * scale;
#pragma unroll
        for (int i = 0; i < 16; ++i) {
            const int hid = (half ? 4 : 0) + ((i & 8) ? 4 : 0) + (i >> 1) + ((i & 1) << 4);
            bfrag[n][i] = (_Float16)(Whh[g * HID + hid] * scale);
        }
    }

    // ---- zero the h tile in LDS (h0 = 0) ---------------------------------
    {
        v8h z = {};
        v8h* p = (v8h*)hbuf;
        p[lane]      = z;
        p[lane + 32] = z;
    }

    // A-fragment LDS element offsets (16-bit A layout, 16x32):
    //  lane<16 : M=lm, K-pos {0..7} and {16..23}
    //  lane>=16: M=lm, K-pos {8..15} and {24..31}
    const int aoff0 = lm * HID + (half ? 8 : 0);
    const int aoff1 = aoff0 + 16;
    // packed h store: element = (j + 8*half)*32 + lm*2 (holds hidden lm, lm+16)
    const int st2 = half * 256 + lm * 2;

    v8f c0 = {}, c1 = {};                // cell state (hid 0-15 / 16-31), C-layout
    v8f h0f = {}, h1f = {};              // last hidden state (f32) for epilogue

    const float4* xq4 = (const float4*)(x + (size_t)(batch_base + lm) * T_STEPS);

    float4 xcur = xq4[0];
    for (int grp = 0; grp < T_STEPS / 4; ++grp) {
        // software-pipeline next 4 timesteps of x (L2-resident; hides latency)
        const int gnext = (grp + 1 < T_STEPS / 4) ? grp + 1 : grp;
        float4 xnext = xq4[gnext];

#pragma unroll
        for (int u = 0; u < 4; ++u) {
            // --- A fragment: previous h from LDS (two b128 loads) ---
            v8h alo = *(const v8h*)(hbuf + aoff0);
            v8h ahi = *(const v8h*)(hbuf + aoff1);
            v16h A;
#pragma unroll
            for (int i = 0; i < 8; ++i) { A[i] = alo[i]; A[8 + i] = ahi[i]; }

            // --- x broadcast: lane lm holds x[batch lm, t]; route to C rows ---
            const float xv = (&xcur.x)[u];
            float xb[8];
#pragma unroll
            for (int j = 0; j < 8; ++j)
                xb[j] = __shfl(xv, j + half * 8, 32);

            // --- 8 WMMAs: D[n] = A x B[n] + (x*wih + bias)  (i/f/o pre-halved) ---
            v8f D[8];
#pragma unroll
            for (int n = 0; n < 8; ++n) {
                v8f C;
#pragma unroll
                for (int j = 0; j < 8; ++j)
                    C[j] = fmaf(xb[j], wihb[n], biasb[n]);
                D[n] = __builtin_amdgcn_wmma_f32_16x16x32_f16(
                    false, A, false, bfrag[n], (short)0, C, false, false);
            }

            // --- gates (i,f,g,o = tiles {0,1},{2,3},{4,5},{6,7}) + state update ---
#pragma unroll
            for (int j = 0; j < 8; ++j) {
                float hv[2];
#pragma unroll
                for (int p = 0; p < 2; ++p) {
                    v8f& c = p ? c1 : c0;
                    const float ig = sigmoid_from_half(D[0 + p][j]);
                    const float fg = sigmoid_from_half(D[2 + p][j]);
                    const float gg = fast_tanh        (D[4 + p][j]);
                    const float og = sigmoid_from_half(D[6 + p][j]);
                    const float cv = fmaf(fg, c[j], ig * gg);
                    c[j] = cv;
                    hv[p] = og * fast_tanh(cv);
                    (p ? h1f : h0f)[j] = hv[p];
                }
                // packed f16 store of (hidden lm, hidden lm+16) for row j+8*half
#if __has_builtin(__builtin_amdgcn_cvt_pkrtz)
                v2h hp = __builtin_bit_cast(
                    v2h, (v2fp16)__builtin_amdgcn_cvt_pkrtz(hv[0], hv[1]));
#else
                v2h hp;
                hp[0] = (_Float16)hv[0];
                hp[1] = (_Float16)hv[1];
#endif
                *(v2h*)(hbuf + st2 + j * HID) = hp;
            }
        }
        xcur = xnext;
    }

    // ---- epilogue: out[b] = h_last . W_lin + b_lin -----------------------
    const float wl0 = Wlin[lm];
    const float wl1 = Wlin[16 + lm];
    const float bl  = blin[0];
#pragma unroll
    for (int j = 0; j < 8; ++j) {
        float partial = h0f[j] * wl0 + h1f[j] * wl1;
        partial += __shfl_xor(partial, 1, 32);
        partial += __shfl_xor(partial, 2, 32);
        partial += __shfl_xor(partial, 4, 32);
        partial += __shfl_xor(partial, 8, 32);
        if (lm == 0)
            out[batch_base + j + half * 8] = partial + bl;
    }
}

extern "C" void kernel_launch(void* const* d_in, const int* in_sizes, int n_in,
                              void* d_out, int out_size, void* d_ws, size_t ws_size,
                              hipStream_t stream) {
    const float* x    = (const float*)d_in[0];
    const float* Wih  = (const float*)d_in[1];
    const float* Whh  = (const float*)d_in[2];
    const float* bih  = (const float*)d_in[3];
    const float* bhh  = (const float*)d_in[4];
    const float* Wlin = (const float*)d_in[5];
    const float* blin = (const float*)d_in[6];
    float* out = (float*)d_out;

    dim3 grid(B_TOTAL / 16);   // 256 independent single-wave workgroups
    dim3 block(32);
    hipLaunchKernelGGL(lstm_wmma_kernel, grid, block, 0, stream,
                       x, Wih, Whh, bih, bhh, Wlin, blin, out);
}
